// HSICLoss_59545426591945
// MI455X (gfx1250) — matrix operand order repaired and checked by reading, not compiled, not run on hardware
//
#include <hip/hip_runtime.h>

// ---------- types ----------
typedef __attribute__((ext_vector_type(16))) __bf16 v16bf;
typedef __attribute__((ext_vector_type(8)))  __bf16 v8bf;
typedef __attribute__((ext_vector_type(8)))  float  v8f;

// vector + address-space types matching the async global->LDS builtin signature:
//   param0: int __vector_size__(16) __device__ (AS1) *
//   param1: int __vector_size__(16) __shared__ (AS3) *   (expected)
typedef int v4i __attribute__((__vector_size__(16)));
typedef __attribute__((address_space(1))) v4i* g_v4i_p;
typedef __attribute__((address_space(3))) v4i* l_v4i_p;

#if defined(__has_builtin)
#if __has_builtin(__builtin_amdgcn_global_load_async_to_lds_b128)
#define HAVE_ASYNC_LDS 1
#endif
#if __has_builtin(__builtin_amdgcn_s_wait_asynccnt)
#define HAVE_WAIT_ASYNC_BUILTIN 1
#endif
#endif

__device__ __forceinline__ void wait_async_zero() {
#if defined(HAVE_ASYNC_LDS)
#if defined(HAVE_WAIT_ASYNC_BUILTIN)
  __builtin_amdgcn_s_wait_asynccnt(0);
#else
  asm volatile("s_wait_asynccnt 0x0" ::: "memory");
#endif
#endif
}

// ---------- problem constants ----------
constexpr int N  = 4096;
constexpr int D  = 512;
constexpr int BT = 64;          // output tile edge (64x64 per block)
constexpr int KC = 64;          // k-chunk staged in LDS per iteration
constexpr int LDK = KC + 8;     // padded LDS row (elements) -> 144B = 9*16B, conflict-free

// workspace layout (bytes)
constexpr size_t XB_OFF  = 0;                           // 4096*512 bf16
constexpr size_t YB_OFF  = XB_OFF + (size_t)N * D * 2;  // 4 MiB
constexpr size_t SQX_OFF = YB_OFF + (size_t)N * D * 2;  // 8 MiB
constexpr size_t SQY_OFF = SQX_OFF + (size_t)N * 4;
constexpr size_t A_OFF   = SQY_OFF + (size_t)N * 4;
constexpr size_t B_OFF   = A_OFF   + (size_t)N * 4;
constexpr size_t S_OFF   = B_OFF   + (size_t)N * 4;

__device__ __forceinline__ unsigned short f32_to_bf16_rne(float f) {
  unsigned int u = __builtin_bit_cast(unsigned int, f);
  unsigned int lsb = (u >> 16) & 1u;
  u += 0x7FFFu + lsb;                       // round to nearest even
  return (unsigned short)(u >> 16);
}
__device__ __forceinline__ float bf16_to_f32(unsigned short h) {
  unsigned int u = ((unsigned int)h) << 16;
  return __builtin_bit_cast(float, u);
}
__device__ __forceinline__ v16bf cat8(v8bf lo, v8bf hi) {
  return __builtin_shufflevector(lo, hi, 0,1,2,3,4,5,6,7,8,9,10,11,12,13,14,15);
}

// one 16-byte tile-row chunk: global -> LDS (async if available)
__device__ __forceinline__ void copy16(const unsigned short* gsrc, unsigned short* ldst) {
#if defined(HAVE_ASYNC_LDS)
  __builtin_amdgcn_global_load_async_to_lds_b128(
      (g_v4i_p)(unsigned short*)gsrc, (l_v4i_p)ldst, /*offset=*/0, /*cpol=*/0);
#else
  *(uint4*)ldst = *(const uint4*)gsrc;
#endif
}

// ---------- kernel 0: zero accumulators (a[4096], b[4096], S) ----------
__global__ void hsic_zero_kernel(float* acc) {
  for (int i = threadIdx.x; i < 2 * N + 1; i += blockDim.x) acc[i] = 0.0f;
}

// ---------- kernel 1: fp32 -> bf16 + row squared norms of bf16-rounded rows ----------
__global__ __launch_bounds__(128)
void hsic_prep_kernel(const float* __restrict__ X, const float* __restrict__ Y,
                      unsigned short* __restrict__ Xb, unsigned short* __restrict__ Yb,
                      float* __restrict__ sqX, float* __restrict__ sqY) {
  const int row = blockIdx.x;
  const float*   src = blockIdx.y ? Y  : X;
  unsigned short* dst = blockIdx.y ? Yb : Xb;
  float*          sq  = blockIdx.y ? sqY : sqX;

  float part = 0.0f;
  for (int c = threadIdx.x; c < D; c += 128) {
    float v = src[(size_t)row * D + c];
    unsigned short h = f32_to_bf16_rne(v);
    dst[(size_t)row * D + c] = h;
    float vb = bf16_to_f32(h);
    part += vb * vb;
  }
  __shared__ float red[128];
  red[threadIdx.x] = part;
  __syncthreads();
  for (int s = 64; s > 0; s >>= 1) {
    if (threadIdx.x < s) red[threadIdx.x] += red[threadIdx.x + s];
    __syncthreads();
  }
  if (threadIdx.x == 0) sq[row] = red[0];
}

// ---------- kernel 2: fused dual-gram WMMA tile + RBF + statistics ----------
__global__ __launch_bounds__(128)
void hsic_tile_kernel(const unsigned short* __restrict__ Xb,
                      const unsigned short* __restrict__ Yb,
                      const float* __restrict__ sqX, const float* __restrict__ sqY,
                      float* __restrict__ aAcc, float* __restrict__ bAcc,
                      float* __restrict__ S) {
  __shared__ unsigned short lXI[BT * LDK];
  __shared__ unsigned short lXJ[BT * LDK];
  __shared__ unsigned short lYI[BT * LDK];
  __shared__ unsigned short lYJ[BT * LDK];
  __shared__ float aRed[BT];
  __shared__ float bRed[BT];
  __shared__ float sRed[128];

  const int i0   = blockIdx.y * BT;   // row block
  const int j0   = blockIdx.x * BT;   // col block
  const int tid  = threadIdx.x;
  const int w    = tid >> 5;          // wave id 0..3: owns rows [w*16, w*16+16)
  const int lane = tid & 31;
  const int lh   = lane & 15;
  const int half = lane >> 4;

  if (tid < BT) { aRed[tid] = 0.0f; bRed[tid] = 0.0f; }

  v8f accX[4] = {};   // 4 col subtiles of the 16x64 strip, gram X
  v8f accY[4] = {};

  for (int ks = 0; ks < D; ks += KC) {
    __syncthreads();  // protect LDS from previous iteration's readers
    // stage 4 tiles of BT x KC bf16; each row = 8 x 16B chunks
    for (int idx = tid; idx < BT * 8; idx += 128) {
      const int r  = idx >> 3;
      const int c8 = (idx & 7) * 8;   // element offset of the 16B chunk
      copy16(Xb + (size_t)(i0 + r) * D + ks + c8, lXI + r * LDK + c8);
      copy16(Xb + (size_t)(j0 + r) * D + ks + c8, lXJ + r * LDK + c8);
      copy16(Yb + (size_t)(i0 + r) * D + ks + c8, lYI + r * LDK + c8);
      copy16(Yb + (size_t)(j0 + r) * D + ks + c8, lYJ + r * LDK + c8);
    }
    wait_async_zero();   // our async copies must land in LDS before we signal
    __syncthreads();

    #pragma unroll
    for (int kk = 0; kk < KC; kk += 32) {
      // A fragment (16x32 bf16): lanes 0-15 -> M=lane, K {0..7,16..23};
      // lanes 16-31 -> M=lane-16, K {8..15,24..31}
      const unsigned short* pAX = lXI + (w * 16 + lh) * LDK + kk + half * 8;
      const unsigned short* pAY = lYI + (w * 16 + lh) * LDK + kk + half * 8;
      v16bf fAX = cat8(*(const v8bf*)pAX, *(const v8bf*)(pAX + 16));
      v16bf fAY = cat8(*(const v8bf*)pAY, *(const v8bf*)(pAY + 16));

      // preload ALL B fragments into distinct registers so the eight WMMAs
      // can issue back-to-back behind one batched s_wait_dscnt
      v16bf fBX[4], fBY[4];
      #pragma unroll
      for (int c = 0; c < 4; ++c) {
        // B fragment (32x16): lane -> N=lane%16, K = (lane<16 ? {0..7,16..23} : {8..15,24..31})
        const unsigned short* pBX = lXJ + (c * 16 + lh) * LDK + kk + half * 8;
        const unsigned short* pBY = lYJ + (c * 16 + lh) * LDK + kk + half * 8;
        fBX[c] = cat8(*(const v8bf*)pBX, *(const v8bf*)(pBX + 16));
        fBY[c] = cat8(*(const v8bf*)pBY, *(const v8bf*)(pBY + 16));
      }
      #pragma unroll
      for (int c = 0; c < 4; ++c) {
        accX[c] = __builtin_amdgcn_wmma_f32_16x16x32_bf16(
            false, fAX, false, fBX[c], (short)0, accX[c], false, false);
        accY[c] = __builtin_amdgcn_wmma_f32_16x16x32_bf16(
            false, fAY, false, fBY[c], (short)0, accY[c], false, false);
      }
    }
  }

  // ---- epilogue: K = exp(G - (sq_i + sq_j)/2) for sigma=1; accumulate stats ----
  // C/D layout: vgpr v, lane L -> M = v + 8*(L>=16), N = L%16
  float sxi[8], syi[8];
  #pragma unroll
  for (int v = 0; v < 8; ++v) {
    const int i = i0 + w * 16 + v + 8 * half;
    sxi[v] = sqX[i];
    syi[v] = sqY[i];
  }

  float rowSumX[8] = {0.f,0.f,0.f,0.f,0.f,0.f,0.f,0.f};
  float rowSumY[8] = {0.f,0.f,0.f,0.f,0.f,0.f,0.f,0.f};
  float sPart = 0.0f;

  #pragma unroll
  for (int c = 0; c < 4; ++c) {
    const int j = j0 + c * 16 + lh;
    const float sxj = sqX[j];
    const float syj = sqY[j];
    #pragma unroll
    for (int v = 0; v < 8; ++v) {
      const float kx = __expf(accX[c][v] - 0.5f * (sxi[v] + sxj));
      const float ky = __expf(accY[c][v] - 0.5f * (syi[v] + syj));
      sPart += kx * ky;
      rowSumX[v] += kx;
      rowSumY[v] += ky;
    }
  }

  __syncthreads();  // aRed/bRed initialized above
  #pragma unroll
  for (int v = 0; v < 8; ++v) {
    const int il = w * 16 + v + 8 * half;
    atomicAdd(&aRed[il], rowSumX[v]);
    atomicAdd(&bRed[il], rowSumY[v]);
  }
  sRed[tid] = sPart;
  __syncthreads();
  for (int s = 64; s > 0; s >>= 1) {
    if (tid < s) sRed[tid] += sRed[tid + s];
    __syncthreads();
  }
  if (tid == 0) atomicAdd(S, sRed[0]);
  if (tid < BT) {
    atomicAdd(&aAcc[i0 + tid], aRed[tid]);
    atomicAdd(&bAcc[i0 + tid], bRed[tid]);
  }
}

// ---------- kernel 3: combine statistics into HSIC scalar ----------
__global__ __launch_bounds__(256)
void hsic_final_kernel(const float* __restrict__ a, const float* __restrict__ b,
                       const float* __restrict__ S, float* __restrict__ out) {
  __shared__ float rab[256], ra[256], rb[256];
  float sab = 0.f, sa = 0.f, sb = 0.f;
  for (int i = threadIdx.x; i < N; i += 256) {
    const float av = a[i], bv = b[i];
    sab += av * bv; sa += av; sb += bv;
  }
  rab[threadIdx.x] = sab; ra[threadIdx.x] = sa; rb[threadIdx.x] = sb;
  __syncthreads();
  for (int s = 128; s > 0; s >>= 1) {
    if (threadIdx.x < s) {
      rab[threadIdx.x] += rab[threadIdx.x + s];
      ra[threadIdx.x]  += ra[threadIdx.x + s];
      rb[threadIdx.x]  += rb[threadIdx.x + s];
    }
    __syncthreads();
  }
  if (threadIdx.x == 0) {
    const float n = (float)N;
    // sum(KXc*KYc) = S - (2/n) * sum(a_i b_i) + (sum a)(sum b)/n^2
    const float total = S[0] - (2.0f / n) * rab[0] + (ra[0] * rb[0]) / (n * n);
    out[0] = total / ((n - 1.0f) * (n - 1.0f));
  }
}

extern "C" void kernel_launch(void* const* d_in, const int* in_sizes, int n_in,
                              void* d_out, int out_size, void* d_ws, size_t ws_size,
                              hipStream_t stream) {
  const float* X = (const float*)d_in[0];
  const float* Y = (const float*)d_in[1];
  float* out = (float*)d_out;
  char* ws = (char*)d_ws;

  unsigned short* Xb  = (unsigned short*)(ws + XB_OFF);
  unsigned short* Yb  = (unsigned short*)(ws + YB_OFF);
  float* sqX = (float*)(ws + SQX_OFF);
  float* sqY = (float*)(ws + SQY_OFF);
  float* aA  = (float*)(ws + A_OFF);
  float* bA  = (float*)(ws + B_OFF);
  float* S   = (float*)(ws + S_OFF);

  hsic_zero_kernel<<<1, 256, 0, stream>>>(aA);  // aA,bA,S are contiguous: 2N+1 floats
  hsic_prep_kernel<<<dim3(N, 2), 128, 0, stream>>>(X, Y, Xb, Yb, sqX, sqY);
  hsic_tile_kernel<<<dim3(N / BT, N / BT), 128, 0, stream>>>(Xb, Yb, sqX, sqY, aA, bA, S);
  hsic_final_kernel<<<1, 256, 0, stream>>>(aA, bA, S, out);
}
